// Model_R_0_34153579938667
// MI455X (gfx1250) — matrix operand order repaired and checked by reading, not compiled, not run on hardware
//
#include <hip/hip_runtime.h>
#include <hip/hip_bf16.h>
#include <math.h>

// ---------------------------------------------------------------------------
// Model: stacked bi/uni LSTMs (WMMA f16 recurrent GEMMs, async-LDS x staging)
// + small conv head. B = N*H*W = 2704 sequences, L = 128 steps.
// ---------------------------------------------------------------------------

typedef __attribute__((ext_vector_type(16))) _Float16 v16h;
typedef __attribute__((ext_vector_type(8)))  _Float16 v8h;
typedef __attribute__((ext_vector_type(8)))  float    v8f;

#define BSEQ 2704
#define LSEQ 128

#define GLOBAL_AS __attribute__((address_space(1)))
#define LDS_AS    __attribute__((address_space(3)))

#if __has_builtin(__builtin_amdgcn_global_load_async_to_lds_b32)
#define HAVE_ASYNC_LDS 1
#else
#define HAVE_ASYNC_LDS 0
#endif

__device__ __forceinline__ void wait_async_zero()
{
#if HAVE_ASYNC_LDS
#if __has_builtin(__builtin_amdgcn_s_wait_asynccnt)
    __builtin_amdgcn_s_wait_asynccnt(0);
#else
    asm volatile("s_wait_asynccnt 0" ::: "memory");
#endif
#endif
}

// gfx1250 native v_tanh_f32 when available
__device__ __forceinline__ float fast_tanh(float x)
{
#if __has_builtin(__builtin_amdgcn_tanhf)
    return __builtin_amdgcn_tanhf(x);
#else
    return tanhf(x);
#endif
}

// --------------------------- traces transpose ------------------------------
// traces [16,128,3,13,13] -> x0 [2704][128][3] (f16), seq = n*169 + h*13 + w
__global__ void transpose_traces_k(const float* __restrict__ traces,
                                   _Float16* __restrict__ x0)
{
    int idx = blockIdx.x * blockDim.x + threadIdx.x;
    const int total = BSEQ * LSEQ * 3;
    if (idx >= total) return;
    int c   = idx % 3;
    int l   = (idx / 3) % LSEQ;
    int seq = idx / (3 * LSEQ);
    int n   = seq / 169;
    int hw  = seq % 169;
    x0[idx] = (_Float16)traces[((n * LSEQ + l) * 3 + c) * 169 + hw];
}

// ------------------------------ LSTM layer ---------------------------------
// One block = 16 sequences, full time recurrence. Gates via WMMA f16:
//   gates[16 x 4H] = [x_t | h][16 x (IN+HID)] @ W^T  + (bih+bhh)
// Double-buffered A staging: async-load x_{t+1} while computing step t.
template<int IN, int HID>
__global__ __launch_bounds__(128)
void lstm_kernel(const _Float16* __restrict__ xin, int xstride, int xcol,
                 _Float16* __restrict__ hout, int ostride, int ocol,
                 const float* __restrict__ Wih, const float* __restrict__ Whh,
                 const float* __restrict__ bih, const float* __restrict__ bhh,
                 int reverse, float* __restrict__ final_out)
{
    constexpr int G  = 4 * HID;              // gate count
    constexpr int KD = IN + HID;             // real K
    constexpr int KP = (KD + 31) / 32 * 32;  // K padded to WMMA chunk
    constexpr int NT = (G + 15) / 16;        // N tiles
    constexpr int GP = NT * 16;              // N padded
    constexpr int KC = KP / 32;              // K chunks

    __shared__ __align__(16) _Float16 sW[GP][KP];      // W^T, N-major: sW[g][k]
    __shared__ __align__(16) _Float16 sA[2][16][KP];   // [x_t | h | 0-pad] x2
    __shared__ float sGate[16][GP];
    __shared__ float sC[16][HID];
    __shared__ float sBias[GP];

    const int tid  = threadIdx.x;
    const int lane = tid & 31;
    const int wave = tid >> 5;
    const int seq0 = blockIdx.x * 16;

    // one-time init: weights (zero-padded), bias, h0=0, c0=0, K-pad of A = 0
    for (int i = tid; i < GP * KP; i += 128) {
        int g = i / KP, k = i % KP;
        float w = 0.f;
        if (g < G) {
            if (k < IN)      w = Wih[g * IN + k];
            else if (k < KD) w = Whh[g * HID + (k - IN)];
        }
        sW[g][k] = (_Float16)w;
    }
    for (int i = tid; i < GP; i += 128)
        sBias[i] = (i < G) ? (bih[i] + bhh[i]) : 0.f;
    for (int i = tid; i < 2 * 16 * KP; i += 128)
        ((_Float16*)sA)[i] = (_Float16)0.f;
    for (int i = tid; i < 16 * HID; i += 128)
        ((float*)sC)[i] = 0.f;
    __syncthreads();

    // stage x_t into buffer pb, cols [0, IN)
    auto stage_x = [&](int t, int pb) {
        if constexpr ((IN & 1) == 0) {
            constexpr int DW = IN / 2;              // dwords per row
            for (int i = tid; i < 16 * DW; i += 128) {
                int r = i / DW, d = (i % DW) * 2;   // element offset (even)
                const _Float16* g =
                    &xin[((seq0 + r) * LSEQ + t) * xstride + xcol + d];
#if HAVE_ASYNC_LDS
                __builtin_amdgcn_global_load_async_to_lds_b32(
                    (GLOBAL_AS int*)(unsigned long long)g,
                    (LDS_AS int*)&sA[pb][r][d],
                    0, 0);
#else
                *(unsigned int*)&sA[pb][r][d] = *(const unsigned int*)g;
#endif
            }
        } else {
            for (int i = tid; i < 16 * IN; i += 128) {
                int r = i / IN, c = i % IN;
                sA[pb][r][c] =
                    xin[((seq0 + r) * LSEQ + t) * xstride + xcol + c];
            }
        }
    };

    // prologue: x for first step into buffer 0
    stage_x(reverse ? (LSEQ - 1) : 0, 0);
    wait_async_zero();
    __syncthreads();

    for (int tt = 0; tt < LSEQ; ++tt) {
        const int t = reverse ? (LSEQ - 1 - tt) : tt;
        const int p = tt & 1;

        // kick off async stage of x_{t+1} into the other buffer
        if (tt + 1 < LSEQ)
            stage_x(reverse ? (LSEQ - 2 - tt) : (tt + 1), 1 - p);

        // gates = A @ W^T  (per-wave N tiles), from buffer p
        for (int tile = wave; tile < NT; tile += 4) {
            v8f acc = {};
            const int m = lane & 15;
            const int n = tile * 16 + m;
            #pragma unroll
            for (int kc = 0; kc < KC; ++kc) {
                // A fragment (16-bit A 16x32): lanes<16 K {0..7,16..23}
                const int ka = kc * 32 + ((lane < 16) ? 0 : 8);
                v8h alo = *(const v8h*)&sA[p][m][ka];
                v8h ahi = *(const v8h*)&sA[p][m][ka + 16];
                // B fragment (16-bit B 32x16): lanes<16 K {0..7,16..23}
                const int kb = kc * 32 + ((lane < 16) ? 0 : 8);
                v8h blo = *(const v8h*)&sW[n][kb];
                v8h bhi = *(const v8h*)&sW[n][kb + 16];
                v16h a, b;
                #pragma unroll
                for (int j = 0; j < 8; ++j) {
                    a[j] = alo[j]; a[j + 8] = ahi[j];
                    b[j] = blo[j]; b[j + 8] = bhi[j];
                }
                acc = __builtin_amdgcn_wmma_f32_16x16x32_f16(
                          false, a, false, b, (short)0, acc, false, false);
            }
            // bias + branchless activation (sigmoid(x)=0.5*tanh(x/2)+0.5)
            const float bias   = sBias[n];
            const bool  isTanh = (n >= 2 * HID) && (n < 3 * HID);  // g gate
            const float sc = isTanh ? 1.f : 0.5f;
            const float aa = isTanh ? 1.f : 0.5f;
            const float bb = isTanh ? 0.f : 0.5f;
            #pragma unroll
            for (int i = 0; i < 8; ++i) {
                const int mrow = i + ((lane < 16) ? 0 : 8);
                float v  = acc[i] + bias;
                float tv = fast_tanh(v * sc);
                sGate[mrow][n] = fmaf(tv, aa, bb);
            }
        }
        __syncthreads();

        // c,h elementwise; feed h into NEXT buffer's cols [IN, IN+HID)
        for (int i = tid; i < 16 * HID; i += 128) {
            const int r = i / HID, j = i % HID;
            const float ig = sGate[r][j];
            const float fg = sGate[r][HID + j];
            const float gg = sGate[r][2 * HID + j];
            const float og = sGate[r][3 * HID + j];
            const float c  = fg * sC[r][j] + ig * gg;
            sC[r][j] = c;
            const float h = og * fast_tanh(c);
            sA[1 - p][r][IN + j] = (_Float16)h;
            hout[((seq0 + r) * LSEQ + t) * ostride + ocol + j] = (_Float16)h;
            if (final_out && tt == LSEQ - 1)
                final_out[(seq0 + r) * HID + j] = h;
        }
        wait_async_zero();   // x_{t+1} landed in LDS
        __syncthreads();
    }
}

// ------------------------------- head --------------------------------------
// feat [16][47][169]: ch 0..9 = u1 last h (NHWC->NCHW), ch 10 = arrival
__global__ void feat0_k(const float* __restrict__ last,
                        const float* __restrict__ arrival,
                        float* __restrict__ feat)
{
    int idx = blockIdx.x * blockDim.x + threadIdx.x;
    const int total = 16 * 11 * 169;
    if (idx >= total) return;
    int p  = idx % 169;
    int ch = (idx / 169) % 11;
    int n  = idx / (169 * 11);
    float v = (ch < 10) ? last[(n * 169 + p) * 10 + ch] : arrival[n * 169 + p];
    feat[(n * 47 + ch) * 169 + p] = v;
}

// generic 1x1 conv: y[n, oc, p] = relu?( sum_ic x[n, ic, p] * w[oc, ic] + b[oc] )
__global__ void conv1x1_k(const float* __restrict__ x, int xcs, int xns, int Cin,
                          const float* __restrict__ w, const float* __restrict__ b,
                          float* __restrict__ y, int ycs, int yns, int Cout,
                          int P, int relu)
{
    int idx = blockIdx.x * blockDim.x + threadIdx.x;
    const int total = 16 * Cout * P;
    if (idx >= total) return;
    int p  = idx % P;
    int oc = (idx / P) % Cout;
    int n  = idx / (P * Cout);
    float s = b[oc];
    for (int ic = 0; ic < Cin; ++ic)
        s += x[n * xns + ic * xcs + p] * w[oc * Cin + ic];
    if (relu) s = s > 0.f ? s : 0.f;
    y[n * yns + oc * ycs + p] = s;
}

// batchnorm stats: one block per channel; stats[2c]=mean, stats[2c+1]=rstd
__global__ void bnstats_k(const float* __restrict__ x, float* __restrict__ stats,
                          int C, int P)
{
    const int c = blockIdx.x;
    float s = 0.f, s2 = 0.f;
    for (int i = threadIdx.x; i < 16 * P; i += blockDim.x) {
        int n = i / P, p = i % P;
        float v = x[(n * C + c) * P + p];
        s += v; s2 += v * v;
    }
    __shared__ float rs[256], rs2[256];
    rs[threadIdx.x] = s; rs2[threadIdx.x] = s2;
    __syncthreads();
    for (int o = 128; o > 0; o >>= 1) {
        if ((int)threadIdx.x < o) {
            rs[threadIdx.x]  += rs[threadIdx.x + o];
            rs2[threadIdx.x] += rs2[threadIdx.x + o];
        }
        __syncthreads();
    }
    if (threadIdx.x == 0) {
        float inv = 1.f / (16.f * (float)P);
        float m   = rs[0] * inv;
        float var = rs2[0] * inv - m * m;
        stats[2 * c]     = m;
        stats[2 * c + 1] = rsqrtf(var + 1e-5f);
    }
}

// y = leaky( gamma*(x-mean)*rstd + beta  [+ res] )
__global__ void bn_act_k(const float* __restrict__ x, const float* __restrict__ stats,
                         const float* __restrict__ gamma, const float* __restrict__ beta,
                         const float* __restrict__ res, float* __restrict__ y,
                         int C, int P)
{
    int idx = blockIdx.x * blockDim.x + threadIdx.x;
    const int total = 16 * C * P;
    if (idx >= total) return;
    int c = (idx / P) % C;
    float v = (x[idx] - stats[2 * c]) * stats[2 * c + 1] * gamma[c] + beta[c];
    if (res) v += res[idx];
    y[idx] = (v >= 0.f) ? v : 0.01f * v;
}

// 2x2 stride-1 average pool: [16][47][13][13] -> [16][47][12][12]
__global__ void avgpool_k(const float* __restrict__ x, float* __restrict__ y)
{
    int idx = blockIdx.x * blockDim.x + threadIdx.x;
    const int total = 16 * 47 * 144;
    if (idx >= total) return;
    int j = idx % 12;
    int i = (idx / 12) % 12;
    int nc = idx / 144;
    const float* xp = x + nc * 169;
    y[idx] = 0.25f * (xp[i * 13 + j] + xp[(i + 1) * 13 + j] +
                      xp[i * 13 + j + 1] + xp[(i + 1) * 13 + j + 1]);
}

// per (n,c) spatial mean over 144 pixels
__global__ void chmean_k(const float* __restrict__ x, float* __restrict__ cm)
{
    int idx = blockIdx.x * blockDim.x + threadIdx.x;
    if (idx >= 16 * 47) return;
    float s = 0.f;
    for (int p = 0; p < 144; ++p) s += x[idx * 144 + p];
    cm[idx] = s * (1.f / 144.f);
}

// out[n, o] = cm[n, :] @ w[o, :] + b[o]
__global__ void fc_k(const float* __restrict__ cm, const float* __restrict__ w,
                     const float* __restrict__ b, float* __restrict__ out, int Cout)
{
    int idx = blockIdx.x * blockDim.x + threadIdx.x;
    if (idx >= 16 * Cout) return;
    int o = idx % Cout;
    int n = idx / Cout;
    float s = b[o];
    for (int c = 0; c < 47; ++c) s += cm[n * 47 + c] * w[o * 47 + c];
    out[n * Cout + o] = s;
}

// ------------------------------- host --------------------------------------
extern "C" void kernel_launch(void* const* d_in, const int* in_sizes, int n_in,
                              void* d_out, int out_size, void* d_ws, size_t ws_size,
                              hipStream_t stream)
{
    (void)in_sizes; (void)n_in; (void)out_size; (void)ws_size;
    const float* traces  = (const float*)d_in[0];
    const float* arrival = (const float*)d_in[1];
    auto F = [&](int i) { return (const float*)d_in[i]; };

    // workspace carve-up (256B aligned)
    char* ws = (char*)d_ws;
    size_t off = 0;
    auto alloc = [&](size_t bytes) -> char* {
        char* r = ws + off;
        off = (off + bytes + 255) & ~(size_t)255;
        return r;
    };
    _Float16* x0  = (_Float16*)alloc((size_t)BSEQ * LSEQ * 3 * 2);
    _Float16* x1  = (_Float16*)alloc((size_t)BSEQ * LSEQ * 100 * 2);
    _Float16* x2  = (_Float16*)alloc((size_t)BSEQ * LSEQ * 100 * 2);
    _Float16* u0  = (_Float16*)alloc((size_t)BSEQ * LSEQ * 10 * 2);
    _Float16* u1  = (_Float16*)alloc((size_t)BSEQ * LSEQ * 10 * 2);
    float* last   = (float*)alloc((size_t)BSEQ * 10 * 4);
    float* feat   = (float*)alloc((size_t)16 * 47 * 169 * 4);
    float* t1     = (float*)alloc((size_t)16 * 47 * 169 * 4);
    float* t2     = (float*)alloc((size_t)16 * 47 * 169 * 4);
    float* y1     = (float*)alloc((size_t)16 * 47 * 169 * 4);
    float* pooled = (float*)alloc((size_t)16 * 47 * 144 * 4);
    float* pt1    = (float*)alloc((size_t)16 * 47 * 144 * 4);
    float* pt2    = (float*)alloc((size_t)16 * 47 * 144 * 4);
    float* py     = (float*)alloc((size_t)16 * 47 * 144 * 4);
    float* stats  = (float*)alloc((size_t)47 * 2 * 4);
    float* cm     = (float*)alloc((size_t)16 * 47 * 4);

    auto g = [](int n) { return dim3((unsigned)((n + 255) / 256)); };

    // 1) input transpose to [seq][t][c] f16
    transpose_traces_k<<<g(BSEQ * LSEQ * 3), 256, 0, stream>>>(traces, x0);

    // 2) LSTM stack (169 blocks of 16 sequences, 4 waves each)
    lstm_kernel<3, 50><<<169, 128, 0, stream>>>(x0, 3, 0, x1, 100, 0,
        F(2), F(3), F(4), F(5), 0, nullptr);
    lstm_kernel<3, 50><<<169, 128, 0, stream>>>(x0, 3, 0, x1, 100, 50,
        F(6), F(7), F(8), F(9), 1, nullptr);
    lstm_kernel<100, 50><<<169, 128, 0, stream>>>(x1, 100, 0, x2, 100, 0,
        F(10), F(11), F(12), F(13), 0, nullptr);
    lstm_kernel<100, 50><<<169, 128, 0, stream>>>(x1, 100, 0, x2, 100, 50,
        F(14), F(15), F(16), F(17), 1, nullptr);
    lstm_kernel<100, 10><<<169, 128, 0, stream>>>(x2, 100, 0, u0, 10, 0,
        F(18), F(19), F(20), F(21), 0, nullptr);
    lstm_kernel<10, 10><<<169, 128, 0, stream>>>(u0, 10, 0, u1, 10, 0,
        F(22), F(23), F(24), F(25), 0, last);

    // 3) feature assembly + concat-conv chain (channels 0..10 -> 47)
    feat0_k<<<g(16 * 11 * 169), 256, 0, stream>>>(last, arrival, feat);
    conv1x1_k<<<g(16 * 12 * 169), 256, 0, stream>>>(feat, 169, 47 * 169, 11,
        F(26), F(27), feat + 11 * 169, 169, 47 * 169, 12, 169, 1);
    conv1x1_k<<<g(16 * 12 * 169), 256, 0, stream>>>(feat, 169, 47 * 169, 23,
        F(28), F(29), feat + 23 * 169, 169, 47 * 169, 12, 169, 1);
    conv1x1_k<<<g(16 * 12 * 169), 256, 0, stream>>>(feat, 169, 47 * 169, 35,
        F(30), F(31), feat + 35 * 169, 169, 47 * 169, 12, 169, 1);

    // 4) task heads
    auto run_res = [&](const float* x, float* y, float* c1, float* c2,
                       const float* const* q, int P) {
        int tot = 16 * 47 * P;
        conv1x1_k<<<g(tot), 256, 0, stream>>>(x, P, 47 * P, 47,
            q[0], q[1], c1, P, 47 * P, 47, P, 0);
        bnstats_k<<<47, 256, 0, stream>>>(c1, stats, 47, P);
        bn_act_k<<<g(tot), 256, 0, stream>>>(c1, stats, q[2], q[3], nullptr, c2, 47, P);
        conv1x1_k<<<g(tot), 256, 0, stream>>>(c2, P, 47 * P, 47,
            q[4], q[5], c1, P, 47 * P, 47, P, 0);
        bnstats_k<<<47, 256, 0, stream>>>(c1, stats, 47, P);
        bn_act_k<<<g(tot), 256, 0, stream>>>(c1, stats, q[6], q[7], x, y, 47, P);
    };
    auto run_task = [&](int base, int Cout, int ooff) {
        const float* q1[8]; const float* q2[8];
        for (int i = 0; i < 8; ++i) { q1[i] = F(base + i); q2[i] = F(base + 8 + i); }
        run_res(feat, y1, t1, t2, q1, 169);
        avgpool_k<<<g(16 * 47 * 144), 256, 0, stream>>>(y1, pooled);
        run_res(pooled, py, pt1, pt2, q2, 144);
        chmean_k<<<g(16 * 47), 256, 0, stream>>>(py, cm);
        fc_k<<<g(16 * Cout), 256, 0, stream>>>(cm, F(base + 16), F(base + 17),
            (float*)d_out + ooff, Cout);
    };
    run_task(32, 1, 0);   // energy  -> out[0:16]
    run_task(50, 2, 16);  // core    -> out[16:48]
    run_task(68, 3, 48);  // axis    -> out[48:96]
    run_task(86, 1, 96);  // xmax    -> out[96:112]
}
